// _GTSDecomposer_62002147885262
// MI455X (gfx1250) — compile-verified, matrix operand order checked by reading, not run on hardware
//
#include <hip/hip_runtime.h>

typedef __attribute__((ext_vector_type(16))) _Float16 v16h;
typedef __attribute__((ext_vector_type(4)))  _Float16 h4;
typedef __attribute__((ext_vector_type(8)))  float    v8f;

#define DD   128
#define EPSV 1e-5f

// 16x32 f16 A/B fragment from LDS (CDNA5 wave32 layout).
// Halves [0..7] = K koff..koff+7 (16 contiguous bytes), halves [8..15] = K 16+koff..23+koff.
// -> exactly two ds_load_b128 per lane.
__device__ __forceinline__ v16h load_frag_lds(const _Float16* __restrict__ base,
                                              int stride, int kbase, int lane) {
  union { v16h v; uint4 q[2]; } f;
  const _Float16* p = base + (lane & 15) * stride + kbase + ((lane & 16) ? 8 : 0);
  f.q[0] = *(const uint4*)(p);
  f.q[1] = *(const uint4*)(p + 16);
  return f.v;
}

__device__ __forceinline__ h4 cvt4(float a, float b, float c, float d) {
  h4 h;
  h[0] = (_Float16)a; h[1] = (_Float16)b; h[2] = (_Float16)c; h[3] = (_Float16)d;
  return h;
}

__global__ void zero_kernel(float* __restrict__ p, long long n) {
  long long i = (long long)blockIdx.x * blockDim.x + threadIdx.x;
  if (i < n) p[i] = 0.f;
}

// Fused: BN -> (128->256) -> ReLU -> (256->256) -> atomic scatter-add.
// One WG = 8 waves, 4 subtiles x 128 token rows (weights staged in LDS once, reused 4x).
__global__ __launch_bounds__(256) void gts_token_kernel(
    const float* __restrict__ x, const int* __restrict__ tok,
    const float* __restrict__ bn_g, const float* __restrict__ bn_b,
    const float* __restrict__ bn_m, const float* __restrict__ bn_v,
    const float* __restrict__ w1, const float* __restrict__ b1,
    const float* __restrict__ w2, const float* __restrict__ b2,
    float* __restrict__ nf, int T_) {
  extern __shared__ char smem[];
  _Float16* sA  = (_Float16*)smem;          // 128 x 128
  _Float16* sW1 = sA  + 128 * 128;          // 256 x 128
  _Float16* sH  = sW1 + 256 * 128;          // 128 x 256
  _Float16* sW2 = sH  + 128 * 256;          // 256 x 256

  const int tid  = threadIdx.x;
  const int lane = tid & 31;
  const int wave = tid >> 5;
  const long long wgBase = (long long)blockIdx.x * 512;

  // Stage weights (f32 -> f16) with 128-bit global loads, 64-bit LDS stores.
  for (int i4 = tid; i4 < (256 * 128) / 4; i4 += 256) {
    float4 w = ((const float4*)w1)[i4];
    ((h4*)sW1)[i4] = cvt4(w.x, w.y, w.z, w.w);
  }
  for (int i4 = tid; i4 < (256 * 256) / 4; i4 += 256) {
    float4 w = ((const float4*)w2)[i4];
    ((h4*)sW2)[i4] = cvt4(w.x, w.y, w.z, w.w);
  }

  // Per-thread BN scale/shift for its invariant 4-column slice of the x tile.
  const int c0 = (tid & 31) * 4;
  float sc[4], sh[4];
#pragma unroll
  for (int j = 0; j < 4; ++j) {
    sc[j] = rsqrtf(bn_v[c0 + j] + EPSV) * bn_g[c0 + j];
    sh[j] = bn_b[c0 + j] - bn_m[c0 + j] * sc[j];
  }

#pragma unroll 1
  for (int st = 0; st < 4; ++st) {
    const long long rb = wgBase + st * 128;

    // Stage BN-normalized x subtile.
    for (int i4 = tid; i4 < 128 * 32; i4 += 256) {
      int r = i4 >> 5;
      long long t = rb + r;
      h4 h = {};
      if (t < T_) {
        float4 v = ((const float4*)x)[t * 32 + (i4 & 31)];
        h = cvt4(v.x * sc[0] + sh[0], v.y * sc[1] + sh[1],
                 v.z * sc[2] + sh[2], v.w * sc[3] + sh[3]);
      }
      ((h4*)sA)[i4] = h;
    }
    __syncthreads();

    // GEMM1: h[128x256] = relu(A @ W1^T + b1); wave owns M-tile `wave`.
    const _Float16* aBase = sA + wave * 16 * 128;
#pragma unroll 1
    for (int n = 0; n < 16; ++n) {
      v8f acc = {};
#pragma unroll
      for (int kc = 0; kc < 4; ++kc) {
        v16h a = load_frag_lds(aBase, 128, kc * 32, lane);
        v16h b = load_frag_lds(sW1 + n * 16 * 128, 128, kc * 32, lane);
        acc = __builtin_amdgcn_wmma_f32_16x16x32_f16(false, a, false, b,
                                                     (short)0, acc, false, false);
      }
#pragma unroll
      for (int r = 0; r < 8; ++r) {
        int row = wave * 16 + r + ((lane & 16) ? 8 : 0);
        int col = n * 16 + (lane & 15);
        float hv = acc[r] + b1[col];
        sH[row * 256 + col] = (_Float16)fmaxf(hv, 0.f);
      }
    }
    // sH rows are private to each wave; no barrier needed between GEMM1 and GEMM2.

    // GEMM2: y[128x256] = H @ W2^T + b2 -> atomic scatter into nf.
    const _Float16* hBase = sH + wave * 16 * 256;
#pragma unroll 1
    for (int n = 0; n < 16; ++n) {
      v8f acc = {};
#pragma unroll
      for (int kc = 0; kc < 8; ++kc) {
        v16h a = load_frag_lds(hBase, 256, kc * 32, lane);
        v16h b = load_frag_lds(sW2 + n * 16 * 256, 256, kc * 32, lane);
        acc = __builtin_amdgcn_wmma_f32_16x16x32_f16(false, a, false, b,
                                                     (short)0, acc, false, false);
      }
#pragma unroll
      for (int r = 0; r < 8; ++r) {
        int rowL = wave * 16 + r + ((lane & 16) ? 8 : 0);
        long long t = rb + rowL;
        int col = n * 16 + (lane & 15);
        if (t < T_) {
          float yv = acc[r] + b2[col];
          int node = (col < DD) ? tok[t] : tok[(long long)T_ + t];
          atomicAdd(&nf[(long long)node * DD + (col & (DD - 1))], yv);
        }
      }
    }
    __syncthreads();   // protect sA before next subtile restages it
  }
}

// Node MLP: BN -> (128->128) -> ReLU -> (128->128). One WG = 4 x 128 node rows.
__global__ __launch_bounds__(256) void gts_node_kernel(
    const float* __restrict__ nf,
    const float* __restrict__ bn_g, const float* __restrict__ bn_b,
    const float* __restrict__ bn_m, const float* __restrict__ bn_v,
    const float* __restrict__ w1, const float* __restrict__ b1,
    const float* __restrict__ w2, const float* __restrict__ b2,
    float* __restrict__ out, int N_) {
  extern __shared__ char smem[];
  _Float16* sA  = (_Float16*)smem;          // 128 x 128
  _Float16* sW1 = sA  + 128 * 128;          // 128 x 128
  _Float16* sH  = sW1 + 128 * 128;          // 128 x 128
  _Float16* sW2 = sH  + 128 * 128;          // 128 x 128

  const int tid  = threadIdx.x;
  const int lane = tid & 31;
  const int wave = tid >> 5;
  const long long wgBase = (long long)blockIdx.x * 512;

  for (int i4 = tid; i4 < (128 * 128) / 4; i4 += 256) {
    float4 wa = ((const float4*)w1)[i4];
    float4 wb = ((const float4*)w2)[i4];
    ((h4*)sW1)[i4] = cvt4(wa.x, wa.y, wa.z, wa.w);
    ((h4*)sW2)[i4] = cvt4(wb.x, wb.y, wb.z, wb.w);
  }

  const int c0 = (tid & 31) * 4;
  float sc[4], sh[4];
#pragma unroll
  for (int j = 0; j < 4; ++j) {
    sc[j] = rsqrtf(bn_v[c0 + j] + EPSV) * bn_g[c0 + j];
    sh[j] = bn_b[c0 + j] - bn_m[c0 + j] * sc[j];
  }

#pragma unroll 1
  for (int st = 0; st < 4; ++st) {
    const long long rb = wgBase + st * 128;

    for (int i4 = tid; i4 < 128 * 32; i4 += 256) {
      int r = i4 >> 5;
      long long nrow = rb + r;
      h4 h = {};
      if (nrow < N_) {
        float4 v = ((const float4*)nf)[nrow * 32 + (i4 & 31)];
        h = cvt4(v.x * sc[0] + sh[0], v.y * sc[1] + sh[1],
                 v.z * sc[2] + sh[2], v.w * sc[3] + sh[3]);
      }
      ((h4*)sA)[i4] = h;
    }
    __syncthreads();

    const _Float16* aBase = sA + wave * 16 * 128;
#pragma unroll 1
    for (int n = 0; n < 8; ++n) {
      v8f acc = {};
#pragma unroll
      for (int kc = 0; kc < 4; ++kc) {
        v16h a = load_frag_lds(aBase, 128, kc * 32, lane);
        v16h b = load_frag_lds(sW1 + n * 16 * 128, 128, kc * 32, lane);
        acc = __builtin_amdgcn_wmma_f32_16x16x32_f16(false, a, false, b,
                                                     (short)0, acc, false, false);
      }
#pragma unroll
      for (int r = 0; r < 8; ++r) {
        int row = wave * 16 + r + ((lane & 16) ? 8 : 0);
        int col = n * 16 + (lane & 15);
        float hv = acc[r] + b1[col];
        sH[row * 128 + col] = (_Float16)fmaxf(hv, 0.f);
      }
    }

    const _Float16* hBase = sH + wave * 16 * 128;
#pragma unroll 1
    for (int n = 0; n < 8; ++n) {
      v8f acc = {};
#pragma unroll
      for (int kc = 0; kc < 4; ++kc) {
        v16h a = load_frag_lds(hBase, 128, kc * 32, lane);
        v16h b = load_frag_lds(sW2 + n * 16 * 128, 128, kc * 32, lane);
        acc = __builtin_amdgcn_wmma_f32_16x16x32_f16(false, a, false, b,
                                                     (short)0, acc, false, false);
      }
#pragma unroll
      for (int r = 0; r < 8; ++r) {
        int rowL = wave * 16 + r + ((lane & 16) ? 8 : 0);
        long long nrow = rb + rowL;
        int col = n * 16 + (lane & 15);
        if (nrow < N_) out[nrow * DD + col] = acc[r] + b2[col];
      }
    }
    __syncthreads();
  }
}

extern "C" void kernel_launch(void* const* d_in, const int* in_sizes, int n_in,
                              void* d_out, int out_size, void* d_ws, size_t ws_size,
                              hipStream_t stream) {
  const float* x     = (const float*)d_in[0];
  const int*   nb    = (const int*)d_in[2];   (void)nb;
  const int*   tok   = (const int*)d_in[3];
  const float* op_g  = (const float*)d_in[4];
  const float* op_b  = (const float*)d_in[5];
  const float* op_m  = (const float*)d_in[6];
  const float* op_v  = (const float*)d_in[7];
  const float* op_w1 = (const float*)d_in[8];
  const float* op_b1 = (const float*)d_in[9];
  const float* op_w2 = (const float*)d_in[10];
  const float* op_b2 = (const float*)d_in[11];
  const float* nm_g  = (const float*)d_in[12];
  const float* nm_b  = (const float*)d_in[13];
  const float* nm_m  = (const float*)d_in[14];
  const float* nm_v  = (const float*)d_in[15];
  const float* nm_w1 = (const float*)d_in[16];
  const float* nm_b1 = (const float*)d_in[17];
  const float* nm_w2 = (const float*)d_in[18];
  const float* nm_b2 = (const float*)d_in[19];

  const int T_ = in_sizes[0] / DD;   // 500000
  const int N_ = in_sizes[2];        // 100000

  float* nf = (float*)d_ws;          // N x 128 f32 accumulator (L2-resident)
  long long nfElems = (long long)N_ * DD;
  zero_kernel<<<(unsigned)((nfElems + 255) / 256), 256, 0, stream>>>(nf, nfElems);

  size_t smemA = (size_t)(128 * 128 + 256 * 128 + 128 * 256 + 256 * 256) * sizeof(_Float16);
  int gridA = (T_ + 511) / 512;
  gts_token_kernel<<<gridA, 256, smemA, stream>>>(
      x, tok, op_g, op_b, op_m, op_v, op_w1, op_b1, op_w2, op_b2, nf, T_);

  size_t smemC = (size_t)(4 * 128 * 128) * sizeof(_Float16);
  int gridC = (N_ + 511) / 512;
  gts_node_kernel<<<gridC, 256, smemC, stream>>>(
      nf, nm_g, nm_b, nm_m, nm_v, nm_w1, nm_b1, nm_w2, nm_b2, (float*)d_out, N_);
}
